// ROAConvModule_43241730736578
// MI455X (gfx1250) — matrix operand order repaired and checked by reading, not compiled, not run on hardware
//
#include <hip/hip_runtime.h>
#include <stdint.h>

typedef __attribute__((ext_vector_type(16))) _Float16     v16h;
typedef __attribute__((ext_vector_type(8)))  float        v8f;
typedef __attribute__((ext_vector_type(4)))  unsigned int v4u;
typedef __attribute__((ext_vector_type(8)))  int          v8i;
typedef __attribute__((ext_vector_type(4)))  int          v4i;

#define BB    16
#define HH    512
#define WW    512
#define TILE  32
#define PAD   4
#define PTILE 40          // TILE + 2*PAD
#define SP    41          // s_tile pitch (odd -> conflict-free rows)
#define CSTR  33          // catbuf row stride (odd -> distinct banks across rows)
#define INV_STD 0.9999950000374997f   // 1/sqrt(1 + 1e-5)
#define DIV_EPS 1e-7f

__device__ __forceinline__ int reflect_idx(int i, int n) {
  i = (i < 0) ? -i : i;
  i = (i > n - 1) ? (2 * (n - 1) - i) : i;
  return i;
}

// Fast ratio via v_rcp_f32 (single-instruction reciprocal, ~1 ulp).
__device__ __forceinline__ float fast_ratio(float num, float den) {
  return num * __builtin_amdgcn_rcpf(den);
}

// Issue one 2D TDM tile DMA: global (wrect x hrect) window of a 512x512 f32
// plane -> contiguous LDS at lds_off. Tracked by TENSORcnt.
__device__ __forceinline__ void tdm_load_2d(uint32_t lds_off, uint64_t gaddr,
                                            int wrect, int hrect) {
  v4u g0;
  g0.x = 1u;                                                // count=1, no gather
  g0.y = lds_off;                                           // LDS byte address
  g0.z = (uint32_t)(gaddr & 0xFFFFFFFFull);                 // global addr lo
  g0.w = (uint32_t)((gaddr >> 32) & 0x1FFFFFFull) | (2u << 30); // hi | type=2
  v8i g1;
  g1[0] = (int)(2u << 16);                                  // data_size = 4B
  g1[1] = (int)((uint32_t)WW << 16);                        // tensor_dim0 lo16
  g1[2] = (int)((((uint32_t)WW) >> 16) | ((uint32_t)HH << 16)); // d0 hi|d1 lo
  g1[3] = (int)((((uint32_t)HH) >> 16) | ((uint32_t)wrect << 16)); // d1 hi|tile_dim0
  g1[4] = (int)((uint32_t)hrect);                           // tile_dim1, tile_dim2=0
  g1[5] = (int)WW;                                          // dim0_stride lo32
  g1[6] = (int)(((uint32_t)(WW * HH) & 0xFFFFu) << 16);     // stride0 hi|stride1 lo
  g1[7] = (int)(((uint32_t)(WW * HH)) >> 16);               // stride1 hi
  v4i g2 = {0, 0, 0, 0};
  v4i g3 = {0, 0, 0, 0};
#if __clang_major__ >= 23
  v8i g4 = {0, 0, 0, 0, 0, 0, 0, 0};
  __builtin_amdgcn_tensor_load_to_lds(g0, g1, g2, g3, g4, 0);
#else
  __builtin_amdgcn_tensor_load_to_lds(g0, g1, g2, g3, 0);
#endif
}

__global__ __launch_bounds__(256) void roaconv_fused(
    const float* __restrict__ x, const float* __restrict__ fuse_w,
    const float* __restrict__ bn_g, const float* __restrict__ bn_b,
    const float* __restrict__ fuse_g, const float* __restrict__ fuse_b,
    float* __restrict__ out) {
  __shared__ float ch[3][PTILE * PTILE];   // raw clamped channel tiles (pitch=wrect)
  __shared__ float s_tile[PTILE * SP];     // reflect-padded channel-sum tile
  __shared__ float catbuf[8][32 * CSTR];   // per-wave 32 pixels x 27(+pad) features

  const int tx0  = blockIdx.x * TILE;
  const int ty0  = blockIdx.y * TILE;
  const int b    = blockIdx.z;
  const int t    = threadIdx.x;
  const int lane = t & 31;
  const int wid  = t >> 5;

  const int y0c = max(ty0 - PAD, 0);
  const int y1c = min(ty0 + TILE - 1 + PAD, HH - 1);
  const int x0c = max(tx0 - PAD, 0);
  const int x1c = min(tx0 + TILE - 1 + PAD, WW - 1);
  const int hrect = y1c - y0c + 1;
  const int wrect = x1c - x0c + 1;

  // ---- Phase 1: TDM async tile DMA, one descriptor per input channel -------
  if (t < 32) {  // wave 0 only (TDM is per-wave, EXEC-ignored)
#pragma unroll
    for (int c = 0; c < 3; ++c) {
      uint64_t g = (uint64_t)(uintptr_t)x +
                   ((((uint64_t)(b * 3 + c)) * HH + (uint64_t)y0c) * WW + (uint64_t)x0c) * 4ull;
      uint32_t lds = (uint32_t)(uintptr_t)(&ch[c][0]);  // generic LDS addr low 32b = offset
      tdm_load_2d(lds, g, wrect, hrect);
    }
    __builtin_amdgcn_s_wait_tensorcnt(0);
  }
  __syncthreads();

  // ---- Phase 2: channel-sum + reflect-pad into s_tile ----------------------
  for (int i = t; i < PTILE * PTILE; i += 256) {
    int py = i / PTILE, px = i - py * PTILE;
    int ry = reflect_idx(ty0 - PAD + py, HH) - y0c;
    int rx = reflect_idx(tx0 - PAD + px, WW) - x0c;
    int idx = ry * wrect + rx;
    s_tile[py * SP + px] = ch[0][idx] + ch[1][idx] + ch[2][idx];
  }
  __syncthreads();

  // ---- A-fragment: fuse_w (4x27) zero-padded to 16x32, f16, built once -----
  // this code's A k order (contiguous, same as B): lanes<16 hold K 0..15, lanes>=16 K 16..31
  v16h afrag;
  {
    const int m = lane & 15;
#pragma unroll
    for (int j = 0; j < 16; ++j) {
      int k = j;
      if (lane >= 16) k += 16;
      float wv = (m < 4 && k < 27) ? fuse_w[m * 27 + k] : 0.f;
      afrag[j] = (_Float16)wv;
    }
  }

  const int xg = tx0 + lane;  // this thread's pixel column

  for (int yy = 0; yy < 4; ++yy) {
    const int lyp = wid * 4 + yy;  // tile-local row
    const int yg  = ty0 + lyp;

    // ---- All 32 directional half-plane sums via row prefix sums ------------
    // acc[half][p-1]: 0=Htop 1=Hbot 2=Vleft 3=Vright 4=D1a(dy>dx) 5=D1b 6=D2a(dy+dx>0) 7=D2b
    float acc[8][4] = {};
#pragma unroll
    for (int r = 0; r < 9; ++r) {
      const float* srow = &s_tile[(lyp + r) * SP + lane];
      float pr[9];
      float a = 0.f;
#pragma unroll
      for (int c = 0; c < 9; ++c) { a += srow[c]; pr[c] = a; }
      auto seg = [&](int c0, int c1) { return pr[c1] - (c0 > 0 ? pr[c0 - 1] : 0.f); };
#pragma unroll
      for (int pi = 0; pi < 4; ++pi) {
        const int p = pi + 1;
        if (r >= 4 - p && r <= 3)     acc[0][pi] += seg(4 - p, 4 + p);
        if (r >= 5     && r <= 4 + p) acc[1][pi] += seg(4 - p, 4 + p);
        if (r >= 4 - p && r <= 4 + p) { acc[2][pi] += seg(4 - p, 3); acc[3][pi] += seg(5, 4 + p); }
        if (r >= 5 - p && r <= 4 + p) acc[4][pi] += seg(4 - p, 4 + ((r - 5) < p ? (r - 5) : p));
        if (r >= 4 - p && r <= 3 + p) acc[5][pi] += seg(4 + ((r - 3) > -p ? (r - 3) : -p), 4 + p);
        if (r >= 5 - p && r <= 4 + p) acc[6][pi] += seg(4 + ((5 - r) > -p ? (5 - r) : -p), 4 + p);
        if (r >= 4 - p && r <= 3 + p) acc[7][pi] += seg(4 - p, 4 + ((3 - r) < p ? (3 - r) : p));
      }
    }

    // ---- ratio + BN affine + gradient magnitudes -> 27-wide feature --------
    float* cb = &catbuf[wid][lane * CSTR];
#pragma unroll
    for (int pi = 0; pi < 4; ++pi) {
      float h  = bn_g[pi]      * INV_STD * fast_ratio(acc[0][pi], acc[1][pi] + DIV_EPS) + bn_b[pi];
      float v  = bn_g[4 + pi]  * INV_STD * fast_ratio(acc[2][pi], acc[3][pi] + DIV_EPS) + bn_b[4 + pi];
      float d1 = bn_g[8 + pi]  * INV_STD * fast_ratio(acc[4][pi], acc[5][pi] + DIV_EPS) + bn_b[8 + pi];
      float d2 = bn_g[12 + pi] * INV_STD * fast_ratio(acc[6][pi], acc[7][pi] + DIV_EPS) + bn_b[12 + pi];
      cb[pi] = h; cb[4 + pi] = v; cb[8 + pi] = d1; cb[12 + pi] = d2;
      cb[16 + pi] = sqrtf(h * h + v * v);
      cb[20 + pi] = sqrtf(d1 * d1 + d2 * d2);
    }
    {
      int idx = (yg - y0c) * wrect + (xg - x0c);
      cb[24] = ch[0][idx]; cb[25] = ch[1][idx]; cb[26] = ch[2][idx];
    }
#pragma unroll
    for (int j = 27; j < 32; ++j) cb[j] = 0.f;  // K padding 27..31
    __builtin_amdgcn_wave_barrier();  // same-wave LDS write->cross-lane read ordering

    // ---- fuse 1x1 conv as v_wmma_f32_16x16x32_f16: 16 pixels per pass ------
    // this code's B k order (contiguous, same as A): lanes<16: N=lane, K 0..15; lanes>=16: N=lane-16, K 16..31
#pragma unroll
    for (int pass = 0; pass < 2; ++pass) {
      const int n   = pass * 16 + (lane & 15);
      const int klo = (lane < 16) ? 0 : 16;
      const float* bp = &catbuf[wid][n * CSTR + klo];
      v16h bfrag;
#pragma unroll
      for (int j = 0; j < 16; ++j) bfrag[j] = (_Float16)bp[j];
      v8f cfr = {0.f, 0.f, 0.f, 0.f, 0.f, 0.f, 0.f, 0.f};
      cfr = __builtin_amdgcn_wmma_f32_16x16x32_f16(false, afrag, false, bfrag,
                                                   (short)0, cfr, false, false);
      if (lane < 16) {  // D rows 0..3 (out channels) live in lanes 0..15, VGPRs 0..3
        const int xo = tx0 + n;
#pragma unroll
        for (int g = 0; g < 4; ++g) {
          float yv = fuse_g[g] * INV_STD * cfr[g] + fuse_b[g];
          // SiLU via fast reciprocal: y * rcp(1 + exp(-y))
          float sv = yv * __builtin_amdgcn_rcpf(1.f + __expf(-yv));
          out[(((size_t)b * 4 + g) * HH + yg) * WW + xo] = sv;
        }
      }
    }
    __builtin_amdgcn_wave_barrier();  // keep next iter's catbuf writes behind reads
  }
}

extern "C" void kernel_launch(void* const* d_in, const int* in_sizes, int n_in,
                              void* d_out, int out_size, void* d_ws, size_t ws_size,
                              hipStream_t stream) {
  const float* x  = (const float*)d_in[0];
  const float* fw = (const float*)d_in[1];
  const float* bg = (const float*)d_in[2];
  const float* bb = (const float*)d_in[3];
  const float* fg = (const float*)d_in[4];
  const float* fb = (const float*)d_in[5];
  dim3 grid(WW / TILE, HH / TILE, BB);
  roaconv_fused<<<grid, 256, 0, stream>>>(x, fw, bg, bb, fg, fb, (float*)d_out);
}
